// BertSelfAttention_61813169324798
// MI455X (gfx1250) — 8-bit matrix operand order repaired and checked by reading, compiled, not run on hardware
//
#include <hip/hip_runtime.h>


typedef __attribute__((ext_vector_type(8))) int v8i;

#define B_   2
#define S_   2048
#define H_   1024
#define NH_  16
#define HD_  64
#define BS_  (B_ * S_)
#define CLIPV 2.5f
#define QMAXV 127.0f

__device__ __forceinline__ void atomicMaxPosF32(float* addr, float v) {
    atomicMax((unsigned int*)addr, __float_as_uint(v));
}

__global__ void init_mx_kernel(float* mx) {
    if (threadIdx.x < 8) mx[threadIdx.x] = 0.0f;
}

__global__ void absmax_clip_kernel(const float* __restrict__ x, long long n,
                                   float* __restrict__ mx_out) {
    __shared__ float sd[256];
    float m = 0.0f;
    for (long long i = (long long)blockIdx.x * 256 + threadIdx.x; i < n;
         i += (long long)gridDim.x * 256) {
        float v = fminf(fmaxf(x[i], -CLIPV), CLIPV);
        m = fmaxf(m, fabsf(v));
    }
    sd[threadIdx.x] = m;
    __syncthreads();
    for (int s = 128; s > 0; s >>= 1) {
        if ((int)threadIdx.x < s) sd[threadIdx.x] = fmaxf(sd[threadIdx.x], sd[threadIdx.x + s]);
        __syncthreads();
    }
    if (threadIdx.x == 0) atomicMaxPosF32(mx_out, sd[0]);
}

__global__ void quant8_kernel(const float* __restrict__ x, signed char* __restrict__ q,
                              long long n, const float* __restrict__ mptr) {
    float m = *mptr;
    float s = QMAXV / m;
    for (long long i = (long long)blockIdx.x * 256 + threadIdx.x; i < n;
         i += (long long)gridDim.x * 256) {
        float v = fminf(fmaxf(x[i], -CLIPV), CLIPV);
        float r = rintf(v * s);
        r = fminf(fmaxf(r, -127.0f), 127.0f);
        q[i] = (signed char)(int)r;
    }
}

__device__ __forceinline__ v8i load_a_frag(const signed char* base, int lda, int k0, int lane) {
    int m = lane & 15, half = lane >> 4;
    const signed char* row = base + (long long)m * lda;
    v8i a;
#pragma unroll
    for (int v = 0; v < 8; ++v) {
        int kk = k0 + (v >> 1) * 16 + (v & 1) * 4 + half * 8;
        a[v] = *(const int*)(row + kk);
    }
    return a;
}
__device__ __forceinline__ v8i load_b_frag(const signed char* base, int ldb, int k0, int lane) {
    int n = lane & 15, half = lane >> 4;
    const signed char* row = base + (long long)n * ldb;
    v8i b;
#pragma unroll
    for (int v = 0; v < 8; ++v) {
        int kk = k0 + (v >> 1) * 16 + (v & 1) * 4 + half * 8;
        b[v] = *(const int*)(row + kk);
    }
    return b;
}

__device__ __forceinline__ v8i wmma_iu8(v8i a, v8i b, v8i c) {
    return __builtin_amdgcn_wmma_i32_16x16x64_iu8(true, a, true, b, c, false, false);
}

__global__ void gemm_qkv_kernel(const signed char* __restrict__ Xq,
                                const signed char* __restrict__ W8,
                                const float* __restrict__ bias,
                                const float* __restrict__ mx, int mxA, int mxW,
                                float* __restrict__ out) {
    int lane = threadIdx.x & 31;
    int wave = threadIdx.x >> 5;
    int tile = blockIdx.x * 8 + wave;
    const int TN = H_ / 32;
    int tr = tile / TN, tc = tile % TN;
    const signed char* A0 = Xq + (long long)tr * 32 * H_;
    const signed char* A1 = A0 + 16 * H_;
    const signed char* B0 = W8 + (long long)tc * 32 * H_;
    const signed char* B1 = B0 + 16 * H_;
    v8i c00 = {0,0,0,0,0,0,0,0}, c01 = {0,0,0,0,0,0,0,0};
    v8i c10 = {0,0,0,0,0,0,0,0}, c11 = {0,0,0,0,0,0,0,0};
#pragma unroll 2
    for (int kt = 0; kt < H_; kt += 64) {
        v8i a0 = load_a_frag(A0, H_, kt, lane);
        v8i a1 = load_a_frag(A1, H_, kt, lane);
        v8i b0 = load_b_frag(B0, H_, kt, lane);
        v8i b1 = load_b_frag(B1, H_, kt, lane);
        c00 = wmma_iu8(a0, b0, c00);
        c01 = wmma_iu8(a0, b1, c01);
        c10 = wmma_iu8(a1, b0, c10);
        c11 = wmma_iu8(a1, b1, c11);
    }
    float s = (mx[mxA] / QMAXV) * (mx[mxW] / QMAXV);
    int n = lane & 15, half = lane >> 4;
    int col0 = tc * 32 + n, col1 = col0 + 16;
    float bv0 = bias[col0], bv1 = bias[col1];
#pragma unroll
    for (int r = 0; r < 8; ++r) {
        int row0 = tr * 32 + r + half * 8;
        int row1 = row0 + 16;
        out[(long long)row0 * H_ + col0] = (float)c00[r] * s + bv0;
        out[(long long)row0 * H_ + col1] = (float)c01[r] * s + bv1;
        out[(long long)row1 * H_ + col0] = (float)c10[r] * s + bv0;
        out[(long long)row1 * H_ + col1] = (float)c11[r] * s + bv1;
    }
}

__global__ void scores_kernel(const signed char* __restrict__ Qq,
                              const signed char* __restrict__ Kq,
                              const float* __restrict__ mask,
                              const float* __restrict__ mx,
                              float* __restrict__ scores) {
    int lane = threadIdx.x & 31;
    int wave = threadIdx.x >> 5;
    long long tile = (long long)blockIdx.x * 8 + wave;
    const int TT = S_ / 32;
    int bh  = (int)(tile / (TT * TT));
    int rem = (int)(tile % (TT * TT));
    int tq = rem / TT, tk = rem % TT;
    int b = bh / NH_, h = bh % NH_;
    const signed char* A0 = Qq + (long long)(b * S_ + tq * 32) * H_ + h * HD_;
    const signed char* A1 = A0 + 16 * H_;
    const signed char* B0 = Kq + (long long)(b * S_ + tk * 32) * H_ + h * HD_;
    const signed char* B1 = B0 + 16 * H_;
    v8i a0 = load_a_frag(A0, H_, 0, lane);
    v8i a1 = load_a_frag(A1, H_, 0, lane);
    v8i b0 = load_b_frag(B0, H_, 0, lane);
    v8i b1 = load_b_frag(B1, H_, 0, lane);
    v8i z = {0,0,0,0,0,0,0,0};
    v8i c00 = wmma_iu8(a0, b0, z);
    v8i c01 = wmma_iu8(a0, b1, z);
    v8i c10 = wmma_iu8(a1, b0, z);
    v8i c11 = wmma_iu8(a1, b1, z);
    float s = (mx[4] / QMAXV) * (mx[5] / QMAXV) * 0.125f;
    int n = lane & 15, half = lane >> 4;
    int kc0 = tk * 32 + n, kc1 = kc0 + 16;
    float mv0 = mask[b * S_ + kc0];
    float mv1 = mask[b * S_ + kc1];
#pragma unroll
    for (int r = 0; r < 8; ++r) {
        long long row0 = (long long)bh * S_ + tq * 32 + r + half * 8;
        long long row1 = row0 + 16;
        scores[row0 * S_ + kc0] = (float)c00[r] * s + mv0;
        scores[row0 * S_ + kc1] = (float)c01[r] * s + mv1;
        scores[row1 * S_ + kc0] = (float)c10[r] * s + mv0;
        scores[row1 * S_ + kc1] = (float)c11[r] * s + mv1;
    }
}

__global__ void softmax_stats_kernel(const float* __restrict__ scores,
                                     float* __restrict__ rowmax,
                                     float* __restrict__ rowsum,
                                     float* __restrict__ mx) {
    __shared__ float sd[256];
    int row = blockIdx.x;
    const float* p = scores + (long long)row * S_;
    int tid = threadIdx.x;
    float m = -3.402823e38f;
    for (int i = tid; i < S_; i += 256) m = fmaxf(m, p[i]);
    sd[tid] = m;
    __syncthreads();
    for (int s = 128; s > 0; s >>= 1) {
        if (tid < s) sd[tid] = fmaxf(sd[tid], sd[tid + s]);
        __syncthreads();
    }
    float rm = sd[0];
    __syncthreads();
    float sum = 0.0f;
    for (int i = tid; i < S_; i += 256) sum += expf(p[i] - rm);
    sd[tid] = sum;
    __syncthreads();
    for (int s = 128; s > 0; s >>= 1) {
        if (tid < s) sd[tid] += sd[tid + s];
        __syncthreads();
    }
    if (tid == 0) {
        rowmax[row] = rm;
        rowsum[row] = sd[0];
        atomicMaxPosF32(mx + 7, 1.0f / sd[0]);
    }
}

__global__ void quant_probs_kernel(const float* __restrict__ scores,
                                   const float* __restrict__ rowmax,
                                   const float* __restrict__ rowsum,
                                   const float* __restrict__ mx,
                                   signed char* __restrict__ Pq) {
    const long long nquad = (long long)B_ * NH_ * S_ * (S_ / 4);
    float qs = QMAXV / mx[7];
    for (long long i = (long long)blockIdx.x * 256 + threadIdx.x; i < nquad;
         i += (long long)gridDim.x * 256) {
        long long base = i * 4;
        int row = (int)(base / S_);
        float rm  = rowmax[row];
        float inv = 1.0f / rowsum[row];
        unsigned int packed = 0;
#pragma unroll
        for (int j = 0; j < 4; ++j) {
            float pr = expf(scores[base + j] - rm) * inv;
            float r  = rintf(pr * qs);
            r = fminf(fmaxf(r, -127.0f), 127.0f);
            packed |= ((unsigned int)(unsigned char)(signed char)(int)r) << (8 * j);
        }
        ((unsigned int*)Pq)[i] = packed;
    }
}

__global__ void transpose_v_kernel(const signed char* __restrict__ Vq,
                                   signed char* __restrict__ Vt) {
    const long long total = (long long)B_ * NH_ * HD_ * S_;
    long long i = (long long)blockIdx.x * 256 + threadIdx.x;
    if (i >= total) return;
    int s  = (int)(i % S_);
    long long t = i / S_;
    int d  = (int)(t % HD_);
    int bh = (int)(t / HD_);
    int b = bh / NH_, h = bh % NH_;
    Vt[i] = Vq[(long long)(b * S_ + s) * H_ + h * HD_ + d];
}

__global__ void ctx_kernel(const signed char* __restrict__ Pq,
                           const signed char* __restrict__ Vt,
                           const float* __restrict__ mx,
                           float* __restrict__ ctx) {
    int lane = threadIdx.x & 31;
    int wave = threadIdx.x >> 5;
    int tile = blockIdx.x * 8 + wave;
    const int TQ = S_ / 32;
    const int TD = HD_ / 32;
    int bh  = tile / (TQ * TD);
    int rem = tile % (TQ * TD);
    int tq = rem / TD, td = rem % TD;
    const signed char* A0 = Pq + ((long long)bh * S_ + tq * 32) * S_;
    const signed char* A1 = A0 + (long long)16 * S_;
    const signed char* B0 = Vt + ((long long)bh * HD_ + td * 32) * S_;
    const signed char* B1 = B0 + (long long)16 * S_;
    v8i c00 = {0,0,0,0,0,0,0,0}, c01 = {0,0,0,0,0,0,0,0};
    v8i c10 = {0,0,0,0,0,0,0,0}, c11 = {0,0,0,0,0,0,0,0};
#pragma unroll 2
    for (int kt = 0; kt < S_; kt += 64) {
        v8i a0 = load_a_frag(A0, S_, kt, lane);
        v8i a1 = load_a_frag(A1, S_, kt, lane);
        v8i b0 = load_b_frag(B0, S_, kt, lane);
        v8i b1 = load_b_frag(B1, S_, kt, lane);
        c00 = wmma_iu8(a0, b0, c00);
        c01 = wmma_iu8(a0, b1, c01);
        c10 = wmma_iu8(a1, b0, c10);
        c11 = wmma_iu8(a1, b1, c11);
    }
    float s = (mx[7] / QMAXV) * (mx[6] / QMAXV);
    int n = lane & 15, half = lane >> 4;
    int b = bh / NH_, h = bh % NH_;
    int d0 = h * HD_ + td * 32 + n, d1 = d0 + 16;
#pragma unroll
    for (int r = 0; r < 8; ++r) {
        int srow0 = tq * 32 + r + half * 8;
        int srow1 = srow0 + 16;
        ctx[(long long)(b * S_ + srow0) * H_ + d0] = (float)c00[r] * s;
        ctx[(long long)(b * S_ + srow0) * H_ + d1] = (float)c01[r] * s;
        ctx[(long long)(b * S_ + srow1) * H_ + d0] = (float)c10[r] * s;
        ctx[(long long)(b * S_ + srow1) * H_ + d1] = (float)c11[r] * s;
    }
}

extern "C" void kernel_launch(void* const* d_in, const int* in_sizes, int n_in,
                              void* d_out, int out_size, void* d_ws, size_t ws_size,
                              hipStream_t stream) {
    (void)in_sizes; (void)n_in; (void)out_size; (void)ws_size;
    const float* x    = (const float*)d_in[0];
    const float* mask = (const float*)d_in[1];
    const float* Wq   = (const float*)d_in[2];
    const float* bq   = (const float*)d_in[3];
    const float* Wk   = (const float*)d_in[4];
    const float* bk   = (const float*)d_in[5];
    const float* Wv   = (const float*)d_in[6];
    const float* bv   = (const float*)d_in[7];

    float* ctx    = (float*)d_out;
    float* scores = (float*)d_out + (long long)B_ * S_ * H_;

    char* ws = (char*)d_ws;
    size_t off = 0;
    auto take = [&](size_t sz) { size_t o = off; off += (sz + 255) & ~(size_t)255; return o; };
    float*       mx     = (float*)(ws + take(32));
    float*       rowmax = (float*)(ws + take((size_t)B_ * NH_ * S_ * 4));
    float*       rowsum = (float*)(ws + take((size_t)B_ * NH_ * S_ * 4));
    signed char* Xq     = (signed char*)(ws + take((size_t)BS_ * H_));
    signed char* Wqq    = (signed char*)(ws + take((size_t)H_ * H_));
    signed char* Wkq    = (signed char*)(ws + take((size_t)H_ * H_));
    signed char* Wvq    = (signed char*)(ws + take((size_t)H_ * H_));
    float*       Qf     = (float*)(ws + take((size_t)BS_ * H_ * 4));
    float*       Kf     = (float*)(ws + take((size_t)BS_ * H_ * 4));
    float*       Vf     = (float*)(ws + take((size_t)BS_ * H_ * 4));
    signed char* Qq     = (signed char*)(ws + take((size_t)BS_ * H_));
    signed char* Kq     = (signed char*)(ws + take((size_t)BS_ * H_));
    signed char* Vq     = (signed char*)(ws + take((size_t)BS_ * H_));
    signed char* Vt     = (signed char*)(ws + take((size_t)BS_ * H_));
    signed char* Pq     = (signed char*)(ws + take((size_t)B_ * NH_ * S_ * S_));

    const long long nX = (long long)BS_ * H_;
    const long long nW = (long long)H_ * H_;

    init_mx_kernel<<<1, 32, 0, stream>>>(mx);

    absmax_clip_kernel<<<1024, 256, 0, stream>>>(x,  nX, mx + 0);
    absmax_clip_kernel<<<256,  256, 0, stream>>>(Wq, nW, mx + 1);
    absmax_clip_kernel<<<256,  256, 0, stream>>>(Wk, nW, mx + 2);
    absmax_clip_kernel<<<256,  256, 0, stream>>>(Wv, nW, mx + 3);

    quant8_kernel<<<2048, 256, 0, stream>>>(x,  Xq,  nX, mx + 0);
    quant8_kernel<<<512,  256, 0, stream>>>(Wq, Wqq, nW, mx + 1);
    quant8_kernel<<<512,  256, 0, stream>>>(Wk, Wkq, nW, mx + 2);
    quant8_kernel<<<512,  256, 0, stream>>>(Wv, Wvq, nW, mx + 3);

    gemm_qkv_kernel<<<512, 256, 0, stream>>>(Xq, Wqq, bq, mx, 0, 1, Qf);
    gemm_qkv_kernel<<<512, 256, 0, stream>>>(Xq, Wkq, bk, mx, 0, 2, Kf);
    gemm_qkv_kernel<<<512, 256, 0, stream>>>(Xq, Wvq, bv, mx, 0, 3, Vf);

    absmax_clip_kernel<<<1024, 256, 0, stream>>>(Qf, nX, mx + 4);
    absmax_clip_kernel<<<1024, 256, 0, stream>>>(Kf, nX, mx + 5);
    absmax_clip_kernel<<<1024, 256, 0, stream>>>(Vf, nX, mx + 6);
    quant8_kernel<<<2048, 256, 0, stream>>>(Qf, Qq, nX, mx + 4);
    quant8_kernel<<<2048, 256, 0, stream>>>(Kf, Kq, nX, mx + 5);
    quant8_kernel<<<2048, 256, 0, stream>>>(Vf, Vq, nX, mx + 6);

    transpose_v_kernel<<<16384, 256, 0, stream>>>(Vq, Vt);

    scores_kernel<<<16384, 256, 0, stream>>>(Qq, Kq, mask, mx, scores);

    softmax_stats_kernel<<<B_ * NH_ * S_, 256, 0, stream>>>(scores, rowmax, rowsum, mx);

    quant_probs_kernel<<<65536, 256, 0, stream>>>(scores, rowmax, rowsum, mx, Pq);

    ctx_kernel<<<512, 256, 0, stream>>>(Pq, Vt, mx, ctx);
}
